// FIM_75316546503221
// MI455X (gfx1250) — matrix operand order repaired and checked by reading, not compiled, not run on hardware
//
#include <hip/hip_runtime.h>

// ---- probes (visible in stderr on any compile failure) ----
#if __has_builtin(__builtin_amdgcn_wmma_f32_16x16x4_f32)
#define USE_F32_WMMA 1
#else
#warning "PROBE: MISSING __builtin_amdgcn_wmma_f32_16x16x4_f32"
#define USE_F32_WMMA 0
#endif
#if __has_builtin(__builtin_amdgcn_global_load_async_to_lds_b128)
#define HAVE_ASYNC 1
#else
#warning "PROBE: MISSING __builtin_amdgcn_global_load_async_to_lds_b128"
#define HAVE_ASYNC 0
#endif

typedef __attribute__((ext_vector_type(2)))  float    v2f;
typedef __attribute__((ext_vector_type(4)))  float    v4f;
typedef __attribute__((ext_vector_type(8)))  float    v8f;
typedef __attribute__((ext_vector_type(4)))  int      v4i;
typedef __attribute__((ext_vector_type(8)))  _Float16 v8h;
typedef __attribute__((ext_vector_type(16))) _Float16 v16h;

#define BATCH 8
#define CCH   64
#define HH    256
#define WW    256
#define TH    8     // output rows per workgroup
#define TW    16    // output pixels per row tile (WMMA N)

// Address-space helpers: AS1 (global) keeps the flat address; AS3 (LDS) is the
// low 32 bits of the generic shared address.
typedef __attribute__((address_space(1))) v4i GV4;
typedef __attribute__((address_space(3))) v4i LV4;
__device__ __forceinline__ GV4* gp128(const void* p) {
    return (GV4*)(unsigned long long)p;
}
__device__ __forceinline__ LV4* lp128(const void* p) {
    return (LV4*)(unsigned)(unsigned long long)p;
}

#if HAVE_ASYNC
#define ASYNC_CP128(lds, g) \
    __builtin_amdgcn_global_load_async_to_lds_b128(gp128(g), lp128(lds), 0, 0)
#define ASYNC_WAIT() asm volatile("s_wait_asynccnt 0x0" ::: "memory")
#else
#define ASYNC_CP128(lds, g) (*(v4f*)(lds) = *(const v4f*)(g))
#define ASYNC_WAIT()
#endif

#if USE_F32_WMMA
// ---------------- exact f32 path: V_WMMA_F32_16X16X4_F32 ----------------
// out[c,p] += sum_i sum_{k=0..23} A_i[c,k] * B_i[k,p]
//   A_i[c,k] = F[c, h0+r+i-3, w0+k-4]          (K axis shifted +1 for 16B DMA)
//   B_i[k,p] = relu(W[i*8+(k-p-1)]) for 0<=k-p-1<8 else 0   (band, k<=23)
__global__ __launch_bounds__(128)
void dynconv_wmma_f32(const float* __restrict__ Mres,
                      const float* __restrict__ Fm,
                      float* __restrict__ Out)
{
    __shared__ float sF[CCH][TH + 7][24];   // 92160 B feature slab (k=0..23)
    __shared__ float sW[64][TH][TW];        // 32768 B raw weights (relu deferred)
    __shared__ float sB[8][6][32][2];       // 12288 B per-lane B fragments

    const int tid = threadIdx.x;
    const int w0  = blockIdx.x * TW;
    const int h0  = blockIdx.y * TH;
    const int b   = blockIdx.z;

    // ---- async DMA: raw weight tile, 4x b128 per (tap,row) ----
    for (int idx = tid; idx < 64 * TH * (TW / 4); idx += 128) {
        const int tap = idx >> 5;
        const int rr  = (idx >> 2) & (TH - 1);
        const int q   = idx & 3;
        ASYNC_CP128(&sW[tap][rr][4 * q],
                    Mres + (((size_t)b * 64 + tap) * HH + (h0 + rr)) * WW + (w0 + 4 * q));
    }
    // ---- async DMA: feature slab, 6x b128 per (ch,row); halo blocks zeroed ----
    for (int rowIdx = tid; rowIdx < CCH * (TH + 7); rowIdx += 128) {
        const int ch  = rowIdx / (TH + 7);
        const int row = rowIdx - ch * (TH + 7);
        const int gh  = h0 + row - 3;
        float* lrow = &sF[ch][row][0];
        if ((unsigned)gh < (unsigned)HH) {
            const float* grow = Fm + (((size_t)b * CCH + ch) * HH + gh) * WW;
            if (w0 > 0) { ASYNC_CP128(&lrow[0], &grow[w0 - 4]); }      // k=0..3
            else        { *(v4f*)&lrow[0] = (v4f){0.f, 0.f, 0.f, 0.f}; }
            #pragma unroll
            for (int q = 0; q < 4; ++q)                                 // k=4..19
                ASYNC_CP128(&lrow[4 + 4 * q], &grow[w0 + 4 * q]);
            if (w0 + 20 <= WW) { ASYNC_CP128(&lrow[20], &grow[w0 + 16]); } // k=20..23
            else               { *(v4f*)&lrow[20] = (v4f){0.f, 0.f, 0.f, 0.f}; }
        } else {
            #pragma unroll
            for (int q = 0; q < 6; ++q)
                *(v4f*)&lrow[4 * q] = (v4f){0.f, 0.f, 0.f, 0.f};
        }
    }
    // zero all B-fragment slots once (band scatter only writes band slots)
    for (int idx = tid; idx < 8 * 6 * 32 * 2; idx += 128)
        ((float*)sB)[idx] = 0.0f;

    ASYNC_WAIT();
    __syncthreads();

    const int lane   = tid & 31;
    const int m      = tid >> 5;       // wave -> channel group
    const int laneHi = lane >> 4;      // 0 or 1
    const int cL     = lane & 15;      // A row (channel in group)
    const int p      = lane & 15;      // B/D column (pixel)

    for (int r = 0; r < TH; ++r) {
        // scatter this row's relu'd weights into per-lane B fragments:
        // K = kk*4 + 2*laneHi + e  ->  kk=k>>2, laneHi=(k>>1)&1, e=k&1
        for (int idx = tid; idx < 8 * 8 * 16; idx += 128) {
            const int i  = idx >> 7;
            const int j  = (idx >> 4) & 7;
            const int pp = idx & 15;
            float v = sW[i * 8 + j][r][pp];
            v = v > 0.0f ? v : 0.0f;
            const int k = pp + j + 1;
            sB[i][k >> 2][(((k >> 1) & 1) << 4) | pp][k & 1] = v;
        }
        __syncthreads();

        v8f acc = {};
        #pragma unroll
        for (int i = 0; i < 8; ++i) {
            const float* fr = &sF[m * 16 + cL][r + i][0];
            #pragma unroll
            for (int kk = 0; kk < 6; ++kk) {            // K = 0..23 covers band
                const int k0 = kk * 4 + 2 * laneHi;     // elem e -> K = k0 + e
                v2f a  = *(const v2f*)&fr[k0];
                v2f bb = *(const v2f*)&sB[i][kk][lane][0];
                acc = __builtin_amdgcn_wmma_f32_16x16x4_f32(
                    false, a, false, bb, (short)0, acc, false, false);
            }
        }

        const size_t base =
            (((size_t)b * CCH + m * 16 + 8 * laneHi) * HH + (h0 + r)) * WW + (w0 + p);
        #pragma unroll
        for (int v = 0; v < 8; ++v)
            Out[base + (size_t)v * HH * WW] = acc[v];
        __syncthreads();    // protect sB before next row's scatter
    }
}

#else // !USE_F32_WMMA ---- f16 fallback: V_WMMA_F32_16X16X32_F16 ----
__global__ __launch_bounds__(128)
void dynconv_wmma_f16(const float* __restrict__ Mres,
                      const float* __restrict__ Fm,
                      float* __restrict__ Out)
{
    __shared__ _Float16 sF[CCH][TH + 7][32];
    __shared__ _Float16 sB[TH][8][32][16];

    const int tid = threadIdx.x;
    const int w0  = blockIdx.x * TW;
    const int h0  = blockIdx.y * TH;
    const int b   = blockIdx.z;

    for (int idx = tid; idx < (int)(sizeof(sB) / 8); idx += 128)
        ((unsigned long long*)sB)[idx] = 0ull;
    for (int idx = tid; idx < CCH * (TH + 7) * 32; idx += 128) {
        const int ch  = idx / ((TH + 7) * 32);
        const int rem = idx - ch * (TH + 7) * 32;
        const int row = rem >> 5;
        const int s   = rem & 31;
        const int gh  = h0 + row - 3;
        const int gw  = w0 + s - 3;
        float v = 0.0f;
        if (s < 23 && (unsigned)gh < (unsigned)HH && (unsigned)gw < (unsigned)WW)
            v = Fm[(((size_t)b * CCH + ch) * HH + gh) * WW + gw];
        sF[ch][row][s] = (_Float16)v;
    }
    __syncthreads();
    for (int idx = tid; idx < TH * 64 * TW; idx += 128) {
        const int r  = idx >> 10;
        const int i  = (idx >> 7) & 7;
        const int j  = (idx >> 4) & 7;
        const int pp = idx & 15;
        float v = Mres[(((size_t)b * 64 + (i * 8 + j)) * HH + (h0 + r)) * WW + (w0 + pp)];
        const int s = pp + j;
        sB[r][i][((s >> 4) << 4) | pp][s & 15] = (_Float16)(v > 0.0f ? v : 0.0f);
    }
    __syncthreads();

    const int lane   = tid & 31;
    const int m      = tid >> 5;
    const int laneHi = lane >> 4;
    const int cL     = lane & 15;
    const int p      = lane & 15;

    for (int r = 0; r < TH; ++r) {
        v8f acc = {};
        #pragma unroll
        for (int i = 0; i < 8; ++i) {
            const v8h* frp = (const v8h*)&sF[m * 16 + cL][r + i][0];
            v16h afrag = __builtin_shufflevector(frp[2 * laneHi], frp[2 * laneHi + 1],
                0, 1, 2, 3, 4, 5, 6, 7, 8, 9, 10, 11, 12, 13, 14, 15);
            const v8h* bp = (const v8h*)&sB[r][i][lane][0];
            v16h bfrag = __builtin_shufflevector(bp[0], bp[1],
                0, 1, 2, 3, 4, 5, 6, 7, 8, 9, 10, 11, 12, 13, 14, 15);
            acc = __builtin_amdgcn_wmma_f32_16x16x32_f16(
                false, afrag, false, bfrag, (short)0, acc, false, false);
        }
        const size_t base =
            (((size_t)b * CCH + m * 16 + 8 * laneHi) * HH + (h0 + r)) * WW + (w0 + p);
        #pragma unroll
        for (int v = 0; v < 8; ++v)
            Out[base + (size_t)v * HH * WW] = acc[v];
    }
}
#endif

extern "C" void kernel_launch(void* const* d_in, const int* in_sizes, int n_in,
                              void* d_out, int out_size, void* d_ws, size_t ws_size,
                              hipStream_t stream) {
    const float* Mres = (const float*)d_in[0];  // (B, 64, H, W) dynamic weights
    const float* Fm   = (const float*)d_in[1];  // (B, 64, H, W) feature map
    float* Out        = (float*)d_out;          // (B, 64, H, W)

    dim3 grid(WW / TW, HH / TH, BATCH);
#if USE_F32_WMMA
    dynconv_wmma_f32<<<grid, 128, 0, stream>>>(Mres, Fm, Out);
#else
    dynconv_wmma_f16<<<grid, 128, 0, stream>>>(Mres, Fm, Out);
#endif
}
